// Prodigy_72164040508155
// MI455X (gfx1250) — matrix operand order repaired and checked by reading, not compiled, not run on hardware
//
#include <hip/hip_runtime.h>
#include <math.h>

#define NN 100000
#define EMB 64
#define HD 32
#define NE 1600000
#define KQVC 192

typedef __attribute__((ext_vector_type(16))) _Float16 v16h;
typedef __attribute__((ext_vector_type(8)))  _Float16 v8h;
typedef __attribute__((ext_vector_type(8)))  float    v8f;

__device__ __forceinline__ v8f wmma_f16(v16h a, v16h b, v8f c) {
  return __builtin_amdgcn_wmma_f32_16x16x32_f16(false, a, false, b, (short)0, c, false, false);
}

// monotonic encoding so float-max == uint-max
__device__ __forceinline__ unsigned enc_f32(float f) {
  unsigned u = __float_as_uint(f);
  return (u & 0x80000000u) ? ~u : (u | 0x80000000u);
}
__device__ __forceinline__ float dec_f32(unsigned u) {
  return __uint_as_float((u & 0x80000000u) ? (u & 0x7FFFFFFFu) : ~u);
}

__global__ void k_zero(unsigned* __restrict__ p, int n) {
  int stride = gridDim.x * blockDim.x;
  for (int i = blockIdx.x * blockDim.x + threadIdx.x; i < n; i += stride) p[i] = 0u;
}

// ---------------- K1: kqv = x @ W_kqv + b_kqv  ([100k,64] x [64,192]) ----------------
__global__ void __launch_bounds__(256) k_kqv(const float* __restrict__ x,
                                             const float* __restrict__ Wk,
                                             const float* __restrict__ bk,
                                             float* __restrict__ kqv) {
  const int lane = threadIdx.x & 31;
  const int gw = (blockIdx.x * blockDim.x + threadIdx.x) >> 5;   // 3000 waves
  const int nt = gw % 12, ms = gw / 12;                          // 12 n-tiles, 250 m-slots
  const int l15 = lane & 15, hsel = lane >> 4;
  const int n0 = nt * 16, col = n0 + l15;

  v16h bfr[2];
  #pragma unroll
  for (int kt = 0; kt < 2; ++kt) {
    int kb = kt * 32 + (hsel << 3);
    #pragma unroll
    for (int j = 0; j < 16; ++j) bfr[kt][j] = (_Float16)Wk[(kb + j + (j & 8)) * KQVC + col];
  }
  const float bias = bk[col];

  for (int mt = ms; mt < 6250; mt += 250) {
    const int m0 = mt * 16;
    const int arow = m0 + l15;
    v8f c = {};
    #pragma unroll
    for (int kt = 0; kt < 2; ++kt) {
      const int kk = kt * 32 + (hsel << 3);
      const float* xp = x + (size_t)arow * EMB + kk;
      v16h a;
      #pragma unroll
      for (int j = 0; j < 8; ++j) { a[j] = (_Float16)xp[j]; a[8 + j] = (_Float16)xp[16 + j]; }
      c = wmma_f16(a, bfr[kt], c);
    }
    const int rbase = m0 + (hsel << 3);
    #pragma unroll
    for (int r = 0; r < 8; ++r)
      kqv[(size_t)(rbase + r) * KQVC + col] = c[r] + bias;
  }
}

// ---------------- K2: per-edge attention scores via WMMA ----------------
// block = 256 threads (8 waves) handles 64 edges = 128 (edge,head) rows.
__global__ void __launch_bounds__(256) k_att(const float* __restrict__ kqv,
                                             const int* __restrict__ ei,
                                             const float* __restrict__ eat,
                                             const float* __restrict__ We,
                                             const float* __restrict__ be,
                                             const float* __restrict__ W1,
                                             const float* __restrict__ b1,
                                             const float* __restrict__ W2,
                                             const float* __restrict__ b2,
                                             float* __restrict__ aout,
                                             unsigned* __restrict__ amax) {
  __shared__ alignas(16) _Float16 feat[128 * 96];
  const int tid = threadIdx.x;
  const int lane = tid & 31;
  const int wid = tid >> 5;
  const int e0 = blockIdx.x * 64;

  // stage feat rows: 2 threads per (edge,head) row
  {
    const int rl = tid >> 1;           // 0..127
    const int s = tid & 1;
    const int e = e0 + (rl >> 1);
    const int h = rl & 1;
    _Float16* frow = feat + rl * 96;
    if (s == 0) {
      const int src = ei[e];
      const float* kp = kqv + (size_t)src * KQVC + EMB + h * HD;  // K block
      const float sc = 0.17677669529663687f;                      // 1/sqrt(32)
      #pragma unroll
      for (int j = 0; j < 32; ++j) frow[j] = (_Float16)(kp[j] * sc);
      const float a0 = eat[(size_t)e * 2 + 0], a1 = eat[(size_t)e * 2 + 1];
      const int cb = h * HD;
      #pragma unroll
      for (int j = 0; j < 32; ++j) {
        float v = a0 * We[cb + j] + a1 * We[64 + cb + j] + be[cb + j];
        frow[64 + j] = (_Float16)fmaxf(v, 0.f);
      }
    } else {
      const int dst = ei[NE + e];
      const float* qp = kqv + (size_t)dst * KQVC + h * HD;        // Q block
      #pragma unroll
      for (int j = 0; j < 32; ++j) frow[32 + j] = (_Float16)qp[j];
    }
  }

  // B fragments of W_att1 [96,32]
  const int l15 = lane & 15, hsel = lane >> 4;
  v16h bfr[3][2];
  #pragma unroll
  for (int kt = 0; kt < 3; ++kt)
    #pragma unroll
    for (int nt = 0; nt < 2; ++nt) {
      const int kb = kt * 32 + (hsel << 3);
      const int cc = nt * 16 + l15;
      #pragma unroll
      for (int j = 0; j < 16; ++j) bfr[kt][nt][j] = (_Float16)W1[(kb + j + (j & 8)) * 32 + cc];
    }
  __syncthreads();

  // A fragments from this wave's 16 rows
  const _Float16* fb = feat + wid * 16 * 96;
  v16h afr[3];
  #pragma unroll
  for (int kt = 0; kt < 3; ++kt) {
    const _Float16* p = fb + l15 * 96 + kt * 32 + (hsel << 3);
    v8h lo = *(const v8h*)p;
    v8h hi = *(const v8h*)(p + 16);
    afr[kt] = __builtin_shufflevector(lo, hi, 0,1,2,3,4,5,6,7,8,9,10,11,12,13,14,15);
  }
  v8f c0 = {}, c1 = {};
  #pragma unroll
  for (int kt = 0; kt < 3; ++kt) c0 = wmma_f16(afr[kt], bfr[kt][0], c0);
  #pragma unroll
  for (int kt = 0; kt < 3; ++kt) c1 = wmma_f16(afr[kt], bfr[kt][1], c1);

  const float bb0 = b1[l15], bb1 = b1[16 + l15];
  const float w2a = W2[l15], w2b = W2[16 + l15];
  float part[8];
  #pragma unroll
  for (int r = 0; r < 8; ++r) {
    float h0 = fmaxf(c0[r] + bb0, 0.f);
    float h1 = fmaxf(c1[r] + bb1, 0.f);
    part[r] = h0 * w2a + h1 * w2b;
  }
  #pragma unroll
  for (int r = 0; r < 8; ++r) {           // reduce over 16 cols (stays within lane-half)
    float v = part[r];
    v += __shfl_xor(v, 1, 32);
    v += __shfl_xor(v, 2, 32);
    v += __shfl_xor(v, 4, 32);
    v += __shfl_xor(v, 8, 32);
    part[r] = v;
  }
  if (l15 == 0) {
    const float bB = b2[0];
    #pragma unroll
    for (int r = 0; r < 8; ++r) {
      const int brow = wid * 16 + (hsel << 3) + r;
      const int e = e0 + (brow >> 1);
      const int h = brow & 1;
      const float av = part[r] + bB;
      aout[(size_t)e * 2 + h] = av;
      const int dst = ei[NE + e];
      atomicMax(&amax[(size_t)dst * 2 + h], enc_f32(av));
    }
  }
}

// ---------------- K3: ex = exp(a - amax[dst]); denom += ex ----------------
__global__ void k_sm(const int* __restrict__ ei, float* __restrict__ aex,
                     const unsigned* __restrict__ amax, float* __restrict__ denom) {
  const int t = blockIdx.x * blockDim.x + threadIdx.x;
  if (t >= NE * 2) return;
  const int e = t >> 1, h = t & 1;
  const int dst = ei[NE + e];
  const float m = dec_f32(amax[(size_t)dst * 2 + h]);
  const float ex = expf(aex[t] - m);
  aex[t] = ex;
  atomicAdd(&denom[(size_t)dst * 2 + h], ex);
}

// ---------------- K4: agg[dst] += alpha * v[src]; deg[dst]++ ----------------
__global__ void k_agg(const int* __restrict__ ei, const float* __restrict__ aex,
                      const float* __restrict__ denom, const float* __restrict__ kqv,
                      float* __restrict__ agg, float* __restrict__ deg) {
  const int t = blockIdx.x * blockDim.x + threadIdx.x;   // 8 threads / edge
  if (t >= NE * 8) return;
  const int e = t >> 3;
  const int c8 = (t & 7) << 3;
  const int h = c8 >> 5;
  const int src = ei[e], dst = ei[NE + e];
  const float alpha = aex[(size_t)e * 2 + h] / denom[(size_t)dst * 2 + h];
  const float* vp = kqv + (size_t)src * KQVC + 128 + c8;
  float* ap = agg + (size_t)dst * EMB + c8;
  #pragma unroll
  for (int j = 0; j < 8; ++j) atomicAdd(ap + j, alpha * vp[j]);
  if ((t & 7) == 0) atomicAdd(&deg[dst], 1.0f);
}

// ---------------- K5: out = BN(agg @ W_out + deg*b_out + x) ----------------
__global__ void __launch_bounds__(256) k_out(const float* __restrict__ agg,
                                             const float* __restrict__ Wo,
                                             const float* __restrict__ bo,
                                             const float* __restrict__ x,
                                             const float* __restrict__ deg,
                                             const float* __restrict__ gamma,
                                             const float* __restrict__ beta,
                                             float* __restrict__ out) {
  const int lane = threadIdx.x & 31;
  const int gw = (blockIdx.x * blockDim.x + threadIdx.x) >> 5;   // 640 waves
  const int l15 = lane & 15, hsel = lane >> 4;

  v16h bfr[2][4];
  #pragma unroll
  for (int kt = 0; kt < 2; ++kt)
    #pragma unroll
    for (int nt = 0; nt < 4; ++nt) {
      const int kb = kt * 32 + (hsel << 3);
      const int cc = nt * 16 + l15;
      #pragma unroll
      for (int j = 0; j < 16; ++j) bfr[kt][nt][j] = (_Float16)Wo[(kb + j + (j & 8)) * EMB + cc];
    }
  const float inv = 0.99999500003749968f;   // 1/sqrt(1 + 1e-5)

  for (int mt = gw; mt < 6250; mt += 640) {
    const int m0 = mt * 16;
    const int arow = m0 + l15;
    v16h afr[2];
    #pragma unroll
    for (int kt = 0; kt < 2; ++kt) {
      const int kk = kt * 32 + (hsel << 3);
      const float* gp = agg + (size_t)arow * EMB + kk;
      #pragma unroll
      for (int j = 0; j < 8; ++j) { afr[kt][j] = (_Float16)gp[j]; afr[kt][8 + j] = (_Float16)gp[16 + j]; }
    }
    #pragma unroll
    for (int nt = 0; nt < 4; ++nt) {
      v8f c = {};
      c = wmma_f16(afr[0], bfr[0][nt], c);
      c = wmma_f16(afr[1], bfr[1][nt], c);
      const int col = nt * 16 + l15;
      const float g = gamma[col] * inv, bt = beta[col], bb = bo[col];
      const int rbase = m0 + (hsel << 3);
      #pragma unroll
      for (int r = 0; r < 8; ++r) {
        const int row = rbase + r;
        const float v = c[r] + deg[row] * bb + x[(size_t)row * EMB + col];
        out[(size_t)row * EMB + col] = v * g + bt;
      }
    }
  }
}

extern "C" void kernel_launch(void* const* d_in, const int* in_sizes, int n_in,
                              void* d_out, int out_size, void* d_ws, size_t ws_size,
                              hipStream_t stream) {
  const float* x   = (const float*)d_in[0];
  const int*   ei  = (const int*)d_in[1];
  const float* eat = (const float*)d_in[2];
  const float* Wk  = (const float*)d_in[3];
  const float* bk  = (const float*)d_in[4];
  const float* We  = (const float*)d_in[5];
  const float* be  = (const float*)d_in[6];
  const float* W1  = (const float*)d_in[7];
  const float* b1  = (const float*)d_in[8];
  const float* W2  = (const float*)d_in[9];
  const float* b2  = (const float*)d_in[10];
  const float* Wo  = (const float*)d_in[11];
  const float* bo  = (const float*)d_in[12];
  const float* gam = (const float*)d_in[13];
  const float* bet = (const float*)d_in[14];

  float* ws = (float*)d_ws;
  float*    kqv   = ws;                                   // 19,200,000 f32
  float*    aex   = kqv + (size_t)NN * KQVC;              //  3,200,000 f32
  unsigned* amax  = (unsigned*)(aex + (size_t)NE * 2);    //    200,000 u32
  float*    denom = (float*)(amax + (size_t)NN * 2);      //    200,000 f32
  float*    agg   = denom + (size_t)NN * 2;               //  6,400,000 f32
  float*    deg   = agg + (size_t)NN * EMB;               //    100,000 f32
  // total ~117.2 MB of workspace

  // zero amax|denom|agg|deg (contiguous: 6,900,000 dwords)
  k_zero<<<4096, 256, 0, stream>>>(amax, 6900000);
  k_kqv<<<375, 256, 0, stream>>>(x, Wk, bk, kqv);                       // 3000 waves
  k_att<<<NE / 64, 256, 0, stream>>>(kqv, ei, eat, We, be, W1, b1, W2, b2, aex, amax);
  k_sm<<<(NE * 2) / 256, 256, 0, stream>>>(ei, aex, amax, denom);
  k_agg<<<(NE * 8) / 256, 256, 0, stream>>>(ei, aex, denom, kqv, agg, deg);
  k_out<<<80, 256, 0, stream>>>(agg, Wo, bo, x, deg, gam, bet, (float*)d_out);
}
